// LocalRelativeMultiheadAttention_31903017074726
// MI455X (gfx1250) — matrix operand order repaired and checked by reading, not compiled, not run on hardware
//
#include <hip/hip_runtime.h>
#include <hip/hip_bf16.h>

// Problem constants (from reference)
#define B_    2
#define S_    2048
#define E_    512
#define H_    8
#define KD_   64
#define VD_   64
#define LB_   128          // LOOK_BACK, LOOK_FORWARD = 0
#define NREL_ 129          // LOOK_BACK + 1 + LOOK_FORWARD

typedef __attribute__((ext_vector_type(16))) __bf16 v16bf;
typedef __attribute__((ext_vector_type(8)))  __bf16 v8bf;
typedef __attribute__((ext_vector_type(8)))  float  v8f;

__device__ __forceinline__ v8f zero8() {
  v8f z;
#pragma unroll
  for (int i = 0; i < 8; ++i) z[i] = 0.0f;
  return z;
}

__device__ __forceinline__ v8f wmma_bf16(v16bf a, v16bf b, v8f c) {
  // D = A(16x32 bf16) * B(32x16 bf16) + C(16x16 f32)
  return __builtin_amdgcn_wmma_f32_16x16x32_bf16(false, a, false, b, (short)0, c,
                                                 false, false);
}

// ---- WMMA fragment loaders (wave32 layouts per CDNA5 ISA 7.12.2) -----------
// A (16x32, 16-bit): lane m=lane&15 holds row m; half=lane>>4 selects
// K=half*8..+7 (vec elems 0..7) and K=16+half*8..+7 (vec elems 8..15).
__device__ __forceinline__ v16bf load_a_frag(const __bf16* base, int ld) {
  int lane = threadIdx.x & 31;
  int m = lane & 15, half = lane >> 4;
  const __bf16* r = base + (size_t)m * ld + half * 8;
  v8bf lo = *(const v8bf*)(r);
  v8bf hi = *(const v8bf*)(r + 16);
  v16bf a;
#pragma unroll
  for (int t = 0; t < 8; ++t) { a[t] = lo[t]; a[8 + t] = hi[t]; }
  return a;
}

// B (32x16, 16-bit): lane n=lane&15 holds column n; half=lane>>4 selects
// K=half*8..+7 (vec elems 0..7) and K=16+half*8..+7 (vec elems 8..15).
// Source layout: src[n*ld + k]  (i.e. "n" rows with k contiguous).
__device__ __forceinline__ v16bf load_b_frag(const __bf16* base, int ld) {
  int lane = threadIdx.x & 31;
  int n = lane & 15, half = lane >> 4;
  const __bf16* r = base + (size_t)n * ld + half * 8;
  v8bf lo = *(const v8bf*)(r);
  v8bf hi = *(const v8bf*)(r + 16);
  v16bf b;
#pragma unroll
  for (int t = 0; t < 8; ++t) { b[t] = lo[t]; b[8 + t] = hi[t]; }
  return b;
}

// B fragment from rel-embedding table with row clamp: B[k][n]=ak[min(n0+n,128)][k0+k]
__device__ __forceinline__ v16bf load_b_frag_rel(const __bf16* ak, int n0, int k0) {
  int lane = threadIdx.x & 31;
  int n = lane & 15, half = lane >> 4;
  int row = n0 + n; if (row > LB_) row = LB_;
  const __bf16* r = ak + row * KD_ + k0 + half * 8;
  v8bf lo = *(const v8bf*)(r);
  v8bf hi = *(const v8bf*)(r + 16);
  v16bf b;
#pragma unroll
  for (int t = 0; t < 8; ++t) { b[t] = lo[t]; b[8 + t] = hi[t]; }
  return b;
}

// ---------------------------------------------------------------------------
// Kernel 0: convert a_k (fp32) -> bf16
__global__ __launch_bounds__(256) void cvt_ak_kernel(const float* __restrict__ ak,
                                                     __bf16* __restrict__ akbf) {
  int t = blockIdx.x * 256 + threadIdx.x;
  if (t < NREL_ * KD_) akbf[t] = (__bf16)ak[t];
}

// ---------------------------------------------------------------------------
// Kernel 1: Q/K/V projections.  grid=(S/64, 3*H, B), block=128 (4 waves).
// Each wave: 16 rows x 64 cols, K-loop over 512 in chunks of 32 via WMMA bf16.
// Q stored PRE-SCALED by 1/sqrt(64) (both score terms are linear in q).
// Q,K stored row-major bf16 [B,H,S,64]; V stored TRANSPOSED bf16 [B,H,64,S].
__global__ __launch_bounds__(128) void proj_kernel(
    const float* __restrict__ xs, const float* __restrict__ wq,
    const float* __restrict__ wk, const float* __restrict__ wv,
    __bf16* __restrict__ qbf, __bf16* __restrict__ kbf,
    __bf16* __restrict__ vtbf) {
  __shared__ __align__(16) __bf16 Wt[64][32];   // Wt[n][k] for current k-chunk

  int lane = threadIdx.x & 31, wave = threadIdx.x >> 5;
  int m = lane & 15, half = lane >> 4;
  int b = blockIdx.z;
  int h = blockIdx.y & 7, proj = blockIdx.y >> 3;
  int i0 = blockIdx.x * 64 + wave * 16;

  const float* W = (proj == 0 ? wq : (proj == 1 ? wk : wv)) + (size_t)h * E_ * KD_;

  v8f acc[4];
#pragma unroll
  for (int nt = 0; nt < 4; ++nt) acc[nt] = zero8();

  for (int e0 = 0; e0 < E_; e0 += 32) {
    // Stage W[e0..e0+31][0..63] into LDS transposed + bf16.
    // Wide row loads (b128), transpose on the LDS-store side.
#pragma unroll
    for (int it = 0; it < 4; ++it) {
      int idx = threadIdx.x + it * 128;            // 0..511 float4s
      int kk = idx >> 4;                           // 0..31  (row e0+kk)
      int nc = (idx & 15) * 4;                     // col start 0..60
      float4 f = *(const float4*)(W + (size_t)(e0 + kk) * KD_ + nc);
      Wt[nc + 0][kk] = (__bf16)f.x;
      Wt[nc + 1][kk] = (__bf16)f.y;
      Wt[nc + 2][kk] = (__bf16)f.z;
      Wt[nc + 3][kk] = (__bf16)f.w;
    }
    __syncthreads();

    // A fragment: xs rows (fp32 -> bf16 on the fly).
    const float* xr = xs + ((size_t)b * S_ + i0 + m) * E_ + e0 + half * 8;
    v16bf a;
#pragma unroll
    for (int t = 0; t < 8; ++t) {
      a[t]     = (__bf16)xr[t];
      a[8 + t] = (__bf16)xr[16 + t];
    }

#pragma unroll
    for (int nt = 0; nt < 4; ++nt) {
      v16bf bf = load_b_frag(&Wt[nt * 16][0], 32);
      acc[nt] = wmma_bf16(a, bf, acc[nt]);
    }
    __syncthreads();
  }

  size_t hb = (size_t)b * H_ + h;
  int n = lane & 15;
  float scl = (proj == 0) ? 0.125f : 1.0f;        // fold 1/sqrt(VD) into Q
  if (proj < 2) {
    __bf16* dst = (proj == 0) ? qbf : kbf;
#pragma unroll
    for (int nt = 0; nt < 4; ++nt)
#pragma unroll
      for (int r = 0; r < 8; ++r) {
        int row = i0 + 8 * half + r;
        dst[(hb * S_ + row) * KD_ + nt * 16 + n] = (__bf16)(acc[nt][r] * scl);
      }
  } else {
#pragma unroll
    for (int nt = 0; nt < 4; ++nt)
#pragma unroll
      for (int r = 0; r < 8; ++r) {
        int row = i0 + 8 * half + r;
        int vc = nt * 16 + n;
        vtbf[(hb * VD_ + vc) * S_ + row] = (__bf16)acc[nt][r];
      }
  }
}

// ---------------------------------------------------------------------------
// Kernel 2: streaming attention.  grid=(S/64, H, B), block=128 (4 waves);
// one wave owns 16 query rows.  Full-row online softmax (all 2048 cols feed
// max/sum), P.V accumulated only for column blocks overlapping the window.
// Q is pre-scaled, so scores are just (q.k + q.a_rel) with no extra mul.
__global__ __launch_bounds__(128) void attn_kernel(
    const __bf16* __restrict__ qbf, const __bf16* __restrict__ kbf,
    const __bf16* __restrict__ vtbf, const __bf16* __restrict__ akbf,
    float* __restrict__ qkv) {
  __shared__ __align__(16) float  Rlds[4][16][144];   // pre-scaled rel scores
  __shared__ __align__(16) __bf16 Plds[4][16][32];    // masked probs (bf16)

  int lane = threadIdx.x & 31, wave = threadIdx.x >> 5;
  int n = lane & 15, half = lane >> 4;
  int b = blockIdx.z, h = blockIdx.y;
  int i0 = blockIdx.x * 64 + wave * 16;

  size_t hb = (size_t)b * H_ + h;
  const __bf16* Q  = qbf  + (hb * S_ + i0) * KD_;
  const __bf16* K  = kbf  + hb * (size_t)S_ * KD_;
  const __bf16* Vt = vtbf + hb * (size_t)VD_ * S_;

  v16bf aq0 = load_a_frag(Q, KD_);        // K-dim 0..31
  v16bf aq1 = load_a_frag(Q + 32, KD_);   // K-dim 32..63

  // R[m][x] = q_scaled_{i0+m} . a_k[min(x,128)],  x in 0..143 (9 tiles)
#pragma unroll
  for (int rt = 0; rt < 9; ++rt) {
    v8f acc = zero8();
    acc = wmma_bf16(aq0, load_b_frag_rel(akbf, rt * 16, 0),  acc);
    acc = wmma_bf16(aq1, load_b_frag_rel(akbf, rt * 16, 32), acc);
#pragma unroll
    for (int r = 0; r < 8; ++r)
      Rlds[wave][8 * half + r][rt * 16 + n] = acc[r];
  }

  float mrun[8], lrun[8];
  v8f o[4];
#pragma unroll
  for (int r = 0; r < 8; ++r) { mrun[r] = -1e30f; lrun[r] = 0.0f; }
#pragma unroll
  for (int vt = 0; vt < 4; ++vt) o[vt] = zero8();

  for (int jb = 0; jb < S_ / 32; ++jb) {
    int j0 = jb * 32;

    // Prefetch next K block (one full 128B row per lane) into the caches.
    __builtin_prefetch(K + (size_t)(j0 + 32) * KD_ + lane * KD_, 0, 1);

    // Raw QK^T for two 16x16 tiles (cols j0..+15 and j0+16..+31).
    v8f s0 = zero8(), s1 = zero8();
    s0 = wmma_bf16(aq0, load_b_frag(K + (size_t)j0 * KD_,             KD_), s0);
    s0 = wmma_bf16(aq1, load_b_frag(K + (size_t)j0 * KD_ + 32,        KD_), s0);
    s1 = wmma_bf16(aq0, load_b_frag(K + (size_t)(j0 + 16) * KD_,      KD_), s1);
    s1 = wmma_bf16(aq1, load_b_frag(K + (size_t)(j0 + 16) * KD_ + 32, KD_), s1);

    float e0v[8], e1v[8];
#pragma unroll
    for (int r = 0; r < 8; ++r) {
      int m = 8 * half + r;
      int d0 = (j0 + n) - (i0 + m);
      int d1 = d0 + 16;
      int idx0 = d0 < -LB_ ? 0 : (d0 > 0 ? LB_ : d0 + LB_);
      int idx1 = d1 < -LB_ ? 0 : (d1 > 0 ? LB_ : d1 + LB_);
      e0v[r] = s0[r] + Rlds[wave][m][idx0];
      e1v[r] = s1[r] + Rlds[wave][m][idx1];
    }

    // Online softmax stats: reductions across the 16 lanes that share a row.
    float mnew[8], alpha[8];
#pragma unroll
    for (int r = 0; r < 8; ++r) {
      float v = fmaxf(e0v[r], e1v[r]);
      v = fmaxf(v, __shfl_xor(v, 1, 16));
      v = fmaxf(v, __shfl_xor(v, 2, 16));
      v = fmaxf(v, __shfl_xor(v, 4, 16));
      v = fmaxf(v, __shfl_xor(v, 8, 16));
      mnew[r]  = fmaxf(mrun[r], v);
      alpha[r] = __expf(mrun[r] - mnew[r]);
      mrun[r]  = mnew[r];
    }

    float p0[8], p1[8];
#pragma unroll
    for (int r = 0; r < 8; ++r) {
      p0[r] = __expf(e0v[r] - mnew[r]);
      p1[r] = __expf(e1v[r] - mnew[r]);
      float s = p0[r] + p1[r];
      s += __shfl_xor(s, 1, 16);
      s += __shfl_xor(s, 2, 16);
      s += __shfl_xor(s, 4, 16);
      s += __shfl_xor(s, 8, 16);
      lrun[r] = lrun[r] * alpha[r] + s;
    }

#pragma unroll
    for (int vt = 0; vt < 4; ++vt)
#pragma unroll
      for (int r = 0; r < 8; ++r) o[vt][r] *= alpha[r];

    // P.V only where the block overlaps the post-softmax window [i-128, i].
    if (j0 >= i0 - 159 && j0 <= i0 + 15) {
#pragma unroll
      for (int r = 0; r < 8; ++r) {
        int m = 8 * half + r;
        int d0 = (j0 + n) - (i0 + m);
        int d1 = d0 + 16;
        Plds[wave][m][n]      = (__bf16)((d0 >= -LB_ && d0 <= 0) ? p0[r] : 0.0f);
        Plds[wave][m][16 + n] = (__bf16)((d1 >= -LB_ && d1 <= 0) ? p1[r] : 0.0f);
      }
      // D-layout -> A-layout relayout through LDS (wave-private, in-order DS).
      v16bf pa = load_a_frag(&Plds[wave][0][0], 32);
#pragma unroll
      for (int vt = 0; vt < 4; ++vt) {
        v16bf bv = load_b_frag(Vt + (size_t)(vt * 16) * S_ + j0, S_);
        o[vt] = wmma_bf16(pa, bv, o[vt]);
      }
    }
  }

  // Normalize by the FULL-row softmax sum; write per-head context.
#pragma unroll
  for (int vt = 0; vt < 4; ++vt)
#pragma unroll
    for (int r = 0; r < 8; ++r) {
      int row = i0 + 8 * half + r;
      qkv[(hb * S_ + row) * VD_ + vt * 16 + n] = o[vt][r] * (1.0f / lrun[r]);
    }
}

// ---------------------------------------------------------------------------
// Kernel 3: out[b,s,v] = sum_h w_o[h] * qkv[b,h,s,v]
__global__ __launch_bounds__(256) void combine_kernel(
    const float* __restrict__ qkv, const float* __restrict__ wo,
    float* __restrict__ out) {
  int tid = blockIdx.x * 256 + threadIdx.x;
  if (tid >= B_ * S_ * VD_) return;
  int v = tid % VD_;
  int s = (tid / VD_) % S_;
  int b = tid / (VD_ * S_);
  float acc = 0.0f;
#pragma unroll
  for (int h = 0; h < H_; ++h)
    acc += wo[h] * qkv[(((size_t)b * H_ + h) * S_ + s) * VD_ + v];
  out[tid] = acc;
}

// ---------------------------------------------------------------------------
extern "C" void kernel_launch(void* const* d_in, const int* in_sizes, int n_in,
                              void* d_out, int out_size, void* d_ws, size_t ws_size,
                              hipStream_t stream) {
  (void)in_sizes; (void)n_in; (void)out_size; (void)ws_size;
  const float* xs = (const float*)d_in[0];
  const float* wq = (const float*)d_in[1];
  const float* wk = (const float*)d_in[2];
  const float* wv = (const float*)d_in[3];
  const float* wo = (const float*)d_in[4];
  const float* ak = (const float*)d_in[5];

  // Workspace layout (~20.03 MB): qbf | kbf | vtbf | akbf | qkv
  const size_t NE = (size_t)B_ * H_ * S_ * KD_;    // 2,097,152 elements
  char* ws = (char*)d_ws;
  __bf16* qbf  = (__bf16*)(ws);                    // 4 MiB
  __bf16* kbf  = (__bf16*)(ws + NE * 2);           // 4 MiB
  __bf16* vtbf = (__bf16*)(ws + NE * 4);           // 4 MiB (transposed V)
  __bf16* akbf = (__bf16*)(ws + NE * 6);           // 16.5 KB
  float*  qkv  = (float*)(ws + NE * 6 + 32768);    // 8 MiB

  cvt_ak_kernel<<<(NREL_ * KD_ + 255) / 256, 256, 0, stream>>>(ak, akbf);
  proj_kernel<<<dim3(S_ / 64, 3 * H_, B_), 128, 0, stream>>>(
      xs, wq, wk, wv, qbf, kbf, vtbf);
  attn_kernel<<<dim3(S_ / 64, H_, B_), 128, 0, stream>>>(
      qbf, kbf, vtbf, akbf, qkv);
  combine_kernel<<<(B_ * S_ * VD_ + 255) / 256, 256, 0, stream>>>(
      qkv, wo, (float*)d_out);
}
